// InnerSoftShiftTripleModule_1589137899673
// MI455X (gfx1250) — compile-verified, matrix operand order checked by reading, not run on hardware
//
#include <hip/hip_runtime.h>
#include <hip/hip_bf16.h>
#include <math.h>

typedef __attribute__((ext_vector_type(16))) _Float16 v16h;
typedef __attribute__((ext_vector_type(8)))  _Float16 v8h;
typedef __attribute__((ext_vector_type(8)))  float    v8f;

#define BATCH   16
#define HW      4096
#define C_IN    512
#define C2      256
#define C_OUT   768
#define M_TOT   1024
#define N_TOT   3072
#define N_TILE  32
#define N_ITERS (N_TOT / N_TILE)   /* 96 */
#define WAVES   8
#define WG_M    (WAVES * 16)       /* 128 M-rows per workgroup */

__device__ __forceinline__ v16h cat16(v8h lo, v8h hi) {
    return __builtin_shufflevector(lo, hi, 0,1,2,3,4,5,6,7,8,9,10,11,12,13,14,15);
}

// ---------------------------------------------------------------------------
// Kernel 1: out[:, 0:512] = input ; out[:, 512:768] = 0  (pure bandwidth)
// ---------------------------------------------------------------------------
__global__ __launch_bounds__(256) void copy_concat_kernel(
        const float* __restrict__ in, float* __restrict__ out) {
    long long i4   = (long long)blockIdx.x * blockDim.x + threadIdx.x;
    long long elem = i4 * 4;                      // out linear index
    int bc = (int)(elem / HW);
    int s  = (int)(elem % HW);
    int b  = bc / C_OUT;
    int c  = bc % C_OUT;
    float4 v;
    if (c < C_IN) {
        v = *(const float4*)(in + ((long long)(b * C_IN + c) * HW + s));
    } else {
        v = make_float4(0.f, 0.f, 0.f, 0.f);
    }
    *(float4*)(out + elem) = v;
}

// ---------------------------------------------------------------------------
// Kernel 2: fused gather + cosine + flash-softmax + PV GEMM + scatter.
// grid = BATCH * (M_TOT / WG_M) = 128 workgroups, 256 threads (8 wave32s).
// Each wave owns 16 M-rows and the full 256-channel output accumulator.
// ---------------------------------------------------------------------------
__global__ __launch_bounds__(256, 1) void soft_shift_attn_kernel(
        const float* __restrict__ in,
        const int*   __restrict__ mask_idx,
        const int*   __restrict__ unmask_idx,
        float*       __restrict__ out) {

    // pl tile, n-major f16 (GEMM1 B fragments: contiguous K runs per n-row)
    __shared__ _Float16 plds[N_TILE][264];        // 264 halves: 16B-aligned rows, bank-skewed
    // pl tile, channel-major f16 (GEMM2 B fragments: contiguous n runs per channel)
    __shared__ _Float16 pldsT[C2][40];            // 40 halves = 80B rows, 16B aligned
    // per-wave staging for P (C-layout -> A-layout re-shuffle)
    __shared__ _Float16 pstage[WAVES][16][40];
    __shared__ float nrm2[N_TILE];
    __shared__ float invn[N_TILE];

    const int tid  = threadIdx.x;
    const int lane = tid & 31;
    const int wave = tid >> 5;
    const int lx   = lane & 15;     // N / M-row index within a 16-wide tile
    const int lg   = lane >> 4;     // lane group: selects K-halves {lg*8.., 16+lg*8..}

    const int b    = blockIdx.x >> 3;
    const int mblk = blockIdx.x & 7;
    const int m_base_wave = mblk * WG_M + wave * 16;

    const float* inb  = in + (long long)b * C_IN * HW;
    const float* latb = inb + (long long)C2 * HW;

    // ---- Load pf A-fragments: 16 rows x K=256, f32 -> f16 in registers ----
    // A layout (16-bit, 16x32): lane lx holds row M=lx; VGPR K-packing
    // {kb..kb+7, kb+16..kb+23} with kb = ks*32 + lg*8.
    v16h a_frag[8];
    {
        const int   m    = m_base_wave + lx;
        const int   pos  = mask_idx[m];
        const float* row = inb + pos;             // element k at row[k*HW]
        #pragma unroll
        for (int s = 0; s < 8; ++s) {
            const int kb = s * 32 + lg * 8;
            v16h f;
            #pragma unroll
            for (int i = 0; i < 8; ++i) {
                f[i]     = (_Float16)row[(long long)(kb + i) * HW];
                f[i + 8] = (_Float16)row[(long long)(kb + 16 + i) * HW];
            }
            a_frag[s] = f;
        }
    }

    // ---- Flash-attention state ----
    float row_max[8], row_sum[8];
    #pragma unroll
    for (int r = 0; r < 8; ++r) { row_max[r] = -__builtin_inff(); row_sum[r] = 0.f; }
    v8f acc[16];
    #pragma unroll
    for (int t = 0; t < 16; ++t) acc[t] = (v8f){};

    const int n_l = tid & 31;      // cooperative-load: position within tile
    const int c0  = tid >> 5;      // cooperative-load: channel phase (0..7)

    for (int nt = 0; nt < N_ITERS; ++nt) {
        __syncthreads();                            // previous tile consumed
        if (tid < N_TILE) nrm2[tid] = 0.f;
        __syncthreads();

        // ---- Cooperative gather of pl tile (coalesced over positions) ----
        {
            const int    pos  = unmask_idx[nt * N_TILE + n_l];
            const float* colp = latb + pos;
            float ss = 0.f;
            #pragma unroll
            for (int i = 0; i < 32; ++i) {
                const int   c = c0 + i * 8;
                const float v = colp[(long long)c * HW];
                ss += v * v;
                const _Float16 hv = (_Float16)v;
                plds[n_l][c]  = hv;
                pldsT[c][n_l] = hv;
            }
            atomicAdd(&nrm2[n_l], ss);              // ds_add_f32
        }
        __syncthreads();
        if (tid < N_TILE) invn[tid] = rsqrtf(nrm2[tid] + 1e-20f);
        __syncthreads();

        // ---- GEMM1: S = pf @ pl^T  (two 16-col subtiles, 8 K-steps each) ----
        v8f S[2];
        #pragma unroll
        for (int j = 0; j < 2; ++j) {
            v8f s = (v8f){};
            const int n = j * 16 + lx;              // B lane holds column n
            #pragma unroll
            for (int ks = 0; ks < 8; ++ks) {
                const int kb = ks * 32 + lg * 8;
                v8h lo = *(const v8h*)(&plds[n][kb]);
                v8h hi = *(const v8h*)(&plds[n][kb + 16]);
                s = __builtin_amdgcn_wmma_f32_16x16x32_f16(
                        false, a_frag[ks], false, cat16(lo, hi),
                        (short)0, s, false, false);
            }
            const float sc = invn[j * 16 + lx];     // deferred 1/||pl_n||
            #pragma unroll
            for (int r = 0; r < 8; ++r) s[r] *= sc;
            S[j] = s;
        }

        // ---- Online softmax (C-layout: row = VGPR r + 8*lg, col = lane lx) ----
        float sf[8];
        #pragma unroll
        for (int r = 0; r < 8; ++r) {
            float tmax = fmaxf(S[0][r], S[1][r]);
            tmax = fmaxf(tmax, __shfl_xor(tmax, 1, 16));
            tmax = fmaxf(tmax, __shfl_xor(tmax, 2, 16));
            tmax = fmaxf(tmax, __shfl_xor(tmax, 4, 16));
            tmax = fmaxf(tmax, __shfl_xor(tmax, 8, 16));
            const float nm = fmaxf(row_max[r], tmax);
            sf[r] = __expf(row_max[r] - nm);
            row_max[r] = nm;
        }
        #pragma unroll
        for (int j = 0; j < 2; ++j)
            #pragma unroll
            for (int r = 0; r < 8; ++r)
                S[j][r] = __expf(S[j][r] - row_max[r]);
        #pragma unroll
        for (int r = 0; r < 8; ++r) {
            float ts = S[0][r] + S[1][r];
            ts += __shfl_xor(ts, 1, 16);
            ts += __shfl_xor(ts, 2, 16);
            ts += __shfl_xor(ts, 4, 16);
            ts += __shfl_xor(ts, 8, 16);
            row_sum[r] = row_sum[r] * sf[r] + ts;
        }
        #pragma unroll
        for (int t = 0; t < 16; ++t)
            #pragma unroll
            for (int r = 0; r < 8; ++r)
                acc[t][r] *= sf[r];

        // ---- Re-layout P: C-layout f32 -> A-layout f16 via per-wave LDS ----
        #pragma unroll
        for (int j = 0; j < 2; ++j)
            #pragma unroll
            for (int r = 0; r < 8; ++r)
                pstage[wave][r + 8 * lg][j * 16 + lx] = (_Float16)S[j][r];
        // intra-wave LDS RAW: DS ops are in-order within a wave
        v16h pf2;
        {
            const int kb = lg * 8;
            v8h lo = *(const v8h*)(&pstage[wave][lx][kb]);
            v8h hi = *(const v8h*)(&pstage[wave][lx][kb + 16]);
            pf2 = cat16(lo, hi);
        }

        // ---- GEMM2: acc[t] += P (16x32) @ pl (32 x 16 channels) ----
        #pragma unroll
        for (int t = 0; t < 16; ++t) {
            const int ch = t * 16 + lx;
            const int kb = lg * 8;
            v8h lo = *(const v8h*)(&pldsT[ch][kb]);
            v8h hi = *(const v8h*)(&pldsT[ch][kb + 16]);
            acc[t] = __builtin_amdgcn_wmma_f32_16x16x32_f16(
                         false, pf2, false, cat16(lo, hi),
                         (short)0, acc[t], false, false);
        }
    }

    // ---- Epilogue: divide by row sums, scatter into out[b, 512+ch, pos_m] ----
    float rcp[8];
    int   posr[8];
    #pragma unroll
    for (int r = 0; r < 8; ++r) {
        rcp[r]  = 1.0f / row_sum[r];
        posr[r] = mask_idx[m_base_wave + r + 8 * lg];
    }
    float* outb = out + (long long)b * C_OUT * HW + (long long)C_IN * HW;
    #pragma unroll
    for (int t = 0; t < 16; ++t) {
        const int ch = t * 16 + lx;
        #pragma unroll
        for (int r = 0; r < 8; ++r)
            outb[(long long)ch * HW + posr[r]] = acc[t][r] * rcp[r];
    }
}

// ---------------------------------------------------------------------------
extern "C" void kernel_launch(void* const* d_in, const int* in_sizes, int n_in,
                              void* d_out, int out_size, void* d_ws, size_t ws_size,
                              hipStream_t stream) {
    (void)in_sizes; (void)n_in; (void)out_size; (void)d_ws; (void)ws_size;
    const float* input      = (const float*)d_in[0];
    const int*   mask_idx   = (const int*)d_in[2];
    const int*   unmask_idx = (const int*)d_in[3];
    float*       out        = (float*)d_out;

    const long long total4 = (long long)BATCH * C_OUT * HW / 4;   // 12,582,912
    copy_concat_kernel<<<dim3((unsigned)(total4 / 256)), dim3(256), 0, stream>>>(input, out);

    soft_shift_attn_kernel<<<dim3(BATCH * (M_TOT / WG_M)), dim3(256), 0, stream>>>(
        input, mask_idx, unmask_idx, out);
}
